// AsymmResBlock_3152505995482
// MI455X (gfx1250) — compile-verified, matrix operand order checked by reading, not run on hardware
//
#include <hip/hip_runtime.h>

// ---------------------------------------------------------------------------
// CDNA5 / gfx1250 WMMA types
// ---------------------------------------------------------------------------
typedef __attribute__((ext_vector_type(16))) __bf16 v16bf;
typedef __attribute__((ext_vector_type(8)))  float  v8f;

// Exact parameter types for __builtin_amdgcn_global_load_async_to_lds_b128:
// (v4i addrspace(1)*, v4i addrspace(3)*, imm int, imm int)  [from hipcc diag]
typedef int v4i __attribute__((vector_size(16)));
typedef v4i __attribute__((address_space(1))) v4i_as1;
typedef v4i __attribute__((address_space(3))) v4i_as3;

#if __has_builtin(__builtin_amdgcn_global_load_async_to_lds_b128)
#define USE_ASYNC_LDS 1
#else
#define USE_ASYNC_LDS 0
#endif

__device__ __forceinline__ unsigned short f2bf(float f) {
  // round-to-nearest-even f32 -> bf16 (storage as ushort)
  unsigned int u = __float_as_uint(f);
  unsigned int r = u + 0x7fffu + ((u >> 16) & 1u);
  return (unsigned short)(r >> 16);
}
__device__ __forceinline__ float bf2f(unsigned short h) {
  return __uint_as_float(((unsigned int)h) << 16);
}

// ---------------------------------------------------------------------------
// Prep: swizzle f32 weights [9, CIN, 128] into bf16 B-fragment order.
// Fragment layout (per 32x16 K x N tile): lane l holds 16 bf16, where
//   N = ntile*16 + (l & 15)
//   K(l, j) = (l>=16)*8 + (j>=8)*16 + (j&7)        (mirrors ISA 16-bit A table)
// Tile order: (k, kc, ntile) -> tile index; 512 bf16 per tile.
// ---------------------------------------------------------------------------
template<int CIN>
__global__ void prep_w_kernel(const float* __restrict__ src,
                              unsigned short* __restrict__ dst) {
  constexpr int KC = CIN / 32;
  int d = blockIdx.x * 256 + threadIdx.x;        // d < 9*KC*8*512 (exact grid)
  int j    =  d        & 15;
  int lane = (d >> 4)  & 31;
  int nt   = (d >> 9)  & 7;
  int kc   = (d >> 12) % KC;
  int k    =  d / (4096 * KC);
  int n = nt * 16 + (lane & 15);
  int c = kc * 32 + ((lane >> 4) << 3) + ((j >> 3) << 4) + (j & 7);
  dst[d] = f2bf(src[(k * CIN + c) * 128 + n]);
}

// BN fold: [gamma,beta,mean,var][128] -> scale[128], shift[128]
__global__ void prep_bn_kernel(const float* __restrict__ p, float* __restrict__ o) {
  int n = threadIdx.x;                            // 128 threads
  float scale = p[n] * rsqrtf(p[384 + n] + 1e-5f);
  o[n] = scale;
  o[128 + n] = p[128 + n] - p[256 + n] * scale;   // beta - mean*scale
}

__global__ void cvt_f32_bf16_kernel(const float* __restrict__ src,
                                    unsigned short* __restrict__ dst, int total) {
  int i = blockIdx.x * 256 + threadIdx.x;
  if (i < total) dst[i] = f2bf(src[i]);
}

// ---------------------------------------------------------------------------
// Sparse submanifold conv layer as gather-GEMM on WMMA bf16.
// Workgroup: 256 thr (8 wave32); tile = 256 voxels x 128 outputs.
// Wave w -> rows [blk*256 + w*32, +32): two 16-row strips, 16 x v8f
// accumulators; every B fragment read from LDS feeds TWO wmmas (B reuse).
// Weights for one tap staged in LDS (KC*8 tiles of 1KB) via async-to-LDS.
// ---------------------------------------------------------------------------
template<int CIN, bool FINAL>
__global__ __launch_bounds__(256)
void subm_conv_wmma(const unsigned short* __restrict__ in,     // bf16 [nvox, CIN]
                    const int*            __restrict__ nbr,    // [9, nvox]
                    const unsigned short* __restrict__ wswz,   // bf16 swizzled
                    const float*          __restrict__ bnss,   // scale[128], shift[128]
                    unsigned short*       __restrict__ outb,   // bf16 [nvox,128]
                    float*                __restrict__ outf,   // f32  [nvox,128] (FINAL)
                    const unsigned short* __restrict__ shortcut, // bf16 (FINAL)
                    int nvox) {
  constexpr int KC     = CIN / 32;
  constexpr int WELEMS = KC * 4096;                 // ushort per tap
  __shared__ unsigned short lds_w[WELEMS];          // 16KB (CIN=64) / 32KB (CIN=128)

  const int tid  = threadIdx.x;
  const int wave = tid >> 5;
  const int lane = tid & 31;
  const int h    = lane >> 4;                       // half-wave select
  const int r    = lane & 15;                       // row within 16-tile
  const int mbase = blockIdx.x * 256 + wave * 32;   // first of this wave's 32 rows
  const int m0 = mbase + r;                         // strip 0 row
  const int m1 = mbase + 16 + r;                    // strip 1 row

  v8f acc[2][8] = {};

  union ABf { v16bf v; uint4 q[2]; };

  for (int k = 0; k < 9; ++k) {
    __syncthreads();                                // protect prior LDS reads
    {
      const uint4* src = (const uint4*)(wswz + k * WELEMS);
      uint4* dst = (uint4*)lds_w;
      #pragma unroll
      for (int i = 0; i < WELEMS / 8 / 256; ++i) {
#if USE_ASYNC_LDS
        // gfx1250 async global->LDS copy, tracked by ASYNCcnt (no VGPR bounce).
        // AS1 value == flat global address; AS3 value == low 32 bits of the
        // generic LDS address (ISA flat->LDS aperture truncation rule).
        __builtin_amdgcn_global_load_async_to_lds_b128(
            (v4i_as1*)(unsigned long long)(uintptr_t)(src + i * 256 + tid),
            (v4i_as3*)(unsigned int)(uintptr_t)(dst + i * 256 + tid),
            0, 0);
#else
        dst[i * 256 + tid] = src[i * 256 + tid];
#endif
      }
#if USE_ASYNC_LDS
#if __has_builtin(__builtin_amdgcn_s_wait_asynccnt)
      __builtin_amdgcn_s_wait_asynccnt(0);
#else
      asm volatile("s_wait_asynccnt 0" ::: "memory");
#endif
#endif
    }
    __syncthreads();

    const int  idx0 = (m0 < nvox) ? nbr[k * nvox + m0] : -1;
    const int  idx1 = (m1 < nvox) ? nbr[k * nvox + m1] : -1;
    const bool act0 = (m0 < nvox) && (idx0 >= 0);
    const bool act1 = (m1 < nvox) && (idx1 >= 0);
    const unsigned short* rowp0 = in + (idx0 < 0 ? 0 : idx0) * CIN;
    const unsigned short* rowp1 = in + (idx1 < 0 ? 0 : idx1) * CIN;

    #pragma unroll
    for (int kc = 0; kc < KC; ++kc) {
      ABf a0, a1;
      if (act0) {
        // ISA 16-bit A layout: lane l -> row l&15; K segs [h*8, h*8+8) and +16
        a0.q[0] = *(const uint4*)(rowp0 + kc * 32 + h * 8);
        a0.q[1] = *(const uint4*)(rowp0 + kc * 32 + 16 + h * 8);
      } else {
        a0.q[0] = make_uint4(0u, 0u, 0u, 0u);
        a0.q[1] = make_uint4(0u, 0u, 0u, 0u);
      }
      if (act1) {
        a1.q[0] = *(const uint4*)(rowp1 + kc * 32 + h * 8);
        a1.q[1] = *(const uint4*)(rowp1 + kc * 32 + 16 + h * 8);
      } else {
        a1.q[0] = make_uint4(0u, 0u, 0u, 0u);
        a1.q[1] = make_uint4(0u, 0u, 0u, 0u);
      }
      #pragma unroll
      for (int nt = 0; nt < 8; ++nt) {
        const v16bf b = *(const v16bf*)(&lds_w[(kc * 8 + nt) * 512 + lane * 16]);
        acc[0][nt] = __builtin_amdgcn_wmma_f32_16x16x32_bf16(
            false, a0.v, false, b, (short)0, acc[0][nt], false, false);
        acc[1][nt] = __builtin_amdgcn_wmma_f32_16x16x32_bf16(
            false, a1.v, false, b, (short)0, acc[1][nt], false, false);
      }
    }
  }

  // Epilogue: LeakyReLU -> BN (folded) -> store (bf16, or f32 + residual).
  // C/D layout: VGPR i, lanes 0-15 -> M=i, N=lane; lanes 16-31 -> M=8+i.
  #pragma unroll
  for (int s = 0; s < 2; ++s) {
    #pragma unroll
    for (int nt = 0; nt < 8; ++nt) {
      const int   n     = nt * 16 + r;
      const float scale = bnss[n];
      const float shift = bnss[128 + n];
      #pragma unroll
      for (int i = 0; i < 8; ++i) {
        const int mo = mbase + s * 16 + h * 8 + i;
        if (mo < nvox) {
          float x = acc[s][nt][i];
          x = (x >= 0.0f) ? x : 0.01f * x;
          float y = x * scale + shift;
          if (FINAL) {
            outf[mo * 128 + n] = y + bf2f(shortcut[mo * 128 + n]);
          } else {
            outb[mo * 128 + n] = f2bf(y);
          }
        }
      }
    }
  }
}

// ---------------------------------------------------------------------------
// Host launcher (graph-capture safe: stream-only launches, ws scratch).
// Inputs: feats, W00, W01, W10, W11, bn00, bn01, bn10, bn11, nbr133, nbr313
// ---------------------------------------------------------------------------
extern "C" void kernel_launch(void* const* d_in, const int* in_sizes, int n_in,
                              void* d_out, int out_size, void* d_ws, size_t ws_size,
                              hipStream_t stream) {
  const float* feats  = (const float*)d_in[0];
  const float* W00    = (const float*)d_in[1];
  const float* W01    = (const float*)d_in[2];
  const float* W10    = (const float*)d_in[3];
  const float* W11    = (const float*)d_in[4];
  const float* bn00   = (const float*)d_in[5];
  const float* bn01   = (const float*)d_in[6];
  const float* bn10   = (const float*)d_in[7];
  const float* bn11   = (const float*)d_in[8];
  const int*   nbr133 = (const int*)d_in[9];
  const int*   nbr313 = (const int*)d_in[10];

  const int N = in_sizes[0] / 64;                  // 200000

  // Workspace layout (bytes)
  char* ws = (char*)d_ws;
  unsigned short* wA   = (unsigned short*)(ws + 0);        // 9*2*4096*2   = 147456
  unsigned short* wB   = (unsigned short*)(ws + 147456);   // 9*4*4096*2   = 294912
  unsigned short* wC   = (unsigned short*)(ws + 442368);   // 147456
  unsigned short* wD   = (unsigned short*)(ws + 589824);   // 294912
  float*          bnS  = (float*)(ws + 884736);            // 4 * 256 * 4  = 4096
  unsigned short* fbf  = (unsigned short*)(ws + 888832);   // N*64*2
  unsigned short* buf0 = (unsigned short*)(ws + 888832 + (size_t)N * 64 * 2);
  unsigned short* buf1 = (unsigned short*)(ws + 888832 + (size_t)N * 64 * 2
                                              + (size_t)N * 128 * 2);

  // --- prep: weight swizzle (bf16 B-fragment order), BN fold, feats->bf16 ---
  prep_w_kernel<64> <<<288, 256, 0, stream>>>(W00, wA);    // 9*2*4096/256
  prep_w_kernel<128><<<576, 256, 0, stream>>>(W01, wB);    // 9*4*4096/256
  prep_w_kernel<64> <<<288, 256, 0, stream>>>(W10, wC);
  prep_w_kernel<128><<<576, 256, 0, stream>>>(W11, wD);
  prep_bn_kernel<<<1, 128, 0, stream>>>(bn00, bnS + 0);
  prep_bn_kernel<<<1, 128, 0, stream>>>(bn01, bnS + 256);
  prep_bn_kernel<<<1, 128, 0, stream>>>(bn10, bnS + 512);
  prep_bn_kernel<<<1, 128, 0, stream>>>(bn11, bnS + 768);
  {
    int total = N * 64;
    cvt_f32_bf16_kernel<<<(total + 255) / 256, 256, 0, stream>>>(feats, fbf, total);
  }

  // --- 4 gather-GEMM layers (buf0 reused after conv0_1 consumes it) ---
  const int cblocks = (N + 255) / 256;
  subm_conv_wmma<64,  false><<<cblocks, 256, 0, stream>>>(
      fbf,  nbr133, wA, bnS + 0,   buf0, nullptr, nullptr, N);          // conv0_0
  subm_conv_wmma<128, false><<<cblocks, 256, 0, stream>>>(
      buf0, nbr313, wB, bnS + 256, buf1, nullptr, nullptr, N);          // conv0_1 -> shortcut
  subm_conv_wmma<64,  false><<<cblocks, 256, 0, stream>>>(
      fbf,  nbr313, wC, bnS + 512, buf0, nullptr, nullptr, N);          // conv1_0
  subm_conv_wmma<128, true><<<cblocks, 256, 0, stream>>>(
      buf0, nbr133, wD, bnS + 768, nullptr, (float*)d_out, buf1, N);    // conv1_1 + residual
}
